// FovealVision_33595234189521
// MI455X (gfx1250) — matrix operand order repaired and checked by reading, not compiled, not run on hardware
//
#include <hip/hip_runtime.h>

#define B_ 2
#define NQ_ 8
#define C_ 256
#define HEADS_ 8
#define HD_ 32
#define HID_ 1024
#define H_ 80
#define W_ 80
#define P_ 7
#define HG_ 94
#define WG_ 94
#define NG_ (HG_*WG_)          // 8836
#define N_ (H_*W_)             // 6400
#define BN_ (B_*NQ_)           // 16
#define ROWS_ (BN_*NG_)        // 141376 (divisible by 16)
#define GT_ ((NG_+15)/16)      // 553
#define TWO_PI 6.283185307179586f

typedef __attribute__((ext_vector_type(16))) __bf16 v16bf;
typedef __attribute__((ext_vector_type(8)))  float  v8f;

__device__ __forceinline__ __bf16 f2bf(float f) {
    unsigned u = __builtin_bit_cast(unsigned, f);
    u = u + 0x7FFFu + ((u >> 16) & 1u);           // round-to-nearest-even
    unsigned short h = (unsigned short)(u >> 16);
    return __builtin_bit_cast(__bf16, h);
}
__device__ __forceinline__ float clampf(float x, float lo, float hi) {
    return fminf(hi, fmaxf(lo, x));
}

// ---------------------------------------------------------------- utilities
__global__ __launch_bounds__(256) void k_fill0(float* p, int n) {
    int i = blockIdx.x * 256 + threadIdx.x;
    if (i < n) p[i] = 0.f;
}
__global__ __launch_bounds__(256) void k_add(const float* a, const float* b, float* o, int n) {
    int i = blockIdx.x * 256 + threadIdx.x;
    if (i < n) o[i] = a[i] + b[i];
}

// ------------------------------------------------- generic small row-GEMM
// out[r][c] = act( sum_k (A[r][k]+Aadd[r][k]) * W[k][c] + bias[c] )
__global__ __launch_bounds__(256) void k_sgemm(const float* A, const float* Aadd,
                                               const float* W, const float* bias,
                                               float* out, int K, int N, int gelu) {
    extern __shared__ float sA[];
    int r = blockIdx.x, tid = threadIdx.x;
    for (int k = tid; k < K; k += 256) {
        float v = A[(size_t)r * K + k];
        if (Aadd) v += Aadd[(size_t)r * K + k];
        sA[k] = v;
    }
    __syncthreads();
    for (int c = tid; c < N; c += 256) {
        float acc = bias ? bias[c] : 0.f;
        for (int k = 0; k < K; ++k) acc += sA[k] * W[(size_t)k * N + c];
        if (gelu) acc = 0.5f * acc * (1.f + erff(acc * 0.70710678118f));
        out[(size_t)r * N + c] = acc;
    }
}

// ------------------------------------- logits + softmax + center/corr/a
__global__ __launch_bounds__(256) void k_fixm(const float* fix, const float* z,
                                              const float* z_pe, float* fixm,
                                              float* out_fixm, float* center,
                                              float* a_log, float* inva, float* out_corr) {
    int bi = blockIdx.x, b = bi / NQ_, tid = threadIdx.x;
    __shared__ float sFix[C_];
    __shared__ float sLog[N_];
    __shared__ float sRed[256];
    for (int c = tid; c < C_; c += 256) sFix[c] = fix[bi * C_ + c];
    __syncthreads();
    float lmax = -1e30f;
    for (int m = tid; m < N_; m += 256) {
        const float* zr = z    + ((size_t)b * N_ + m) * C_;
        const float* pr = z_pe + ((size_t)b * N_ + m) * C_;
        float acc = 0.f;
        for (int c = 0; c < C_; ++c) acc += sFix[c] * (zr[c] + pr[c]);
        sLog[m] = acc;
        lmax = fmaxf(lmax, acc);
    }
    sRed[tid] = lmax; __syncthreads();
    for (int s = 128; s > 0; s >>= 1) { if (tid < s) sRed[tid] = fmaxf(sRed[tid], sRed[tid + s]); __syncthreads(); }
    float mx = sRed[0]; __syncthreads();
    float ls = 0.f, sx = 0.f, sy = 0.f;
    for (int m = tid; m < N_; m += 256) {
        float e = expf(sLog[m] - mx);
        sLog[m] = e; ls += e;
        int y = m / W_, x = m % W_;
        sx += e * (-1.f + 2.f * (float)(x + P_) / (float)(WG_ - 1));
        sy += e * (-1.f + 2.f * (float)(y + P_) / (float)(HG_ - 1));
    }
    sRed[tid] = ls; __syncthreads();
    for (int s = 128; s > 0; s >>= 1) { if (tid < s) sRed[tid] += sRed[tid + s]; __syncthreads(); }
    float S = sRed[0]; __syncthreads();
    sRed[tid] = sx; __syncthreads();
    for (int s = 128; s > 0; s >>= 1) { if (tid < s) sRed[tid] += sRed[tid + s]; __syncthreads(); }
    float SX = sRed[0]; __syncthreads();
    sRed[tid] = sy; __syncthreads();
    for (int s = 128; s > 0; s >>= 1) { if (tid < s) sRed[tid] += sRed[tid + s]; __syncthreads(); }
    float SY = sRed[0]; __syncthreads();
    float inv = 1.f / S;
    for (int m = tid; m < N_; m += 256) {
        float v = sLog[m] * inv;
        int y = m / W_, x = m % W_;
        fixm[(size_t)bi * NG_ + (y + P_) * WG_ + (x + P_)] = v;
        out_fixm[(size_t)bi * N_ + m] = v;
    }
    if (tid == 0) {
        float cx = clampf(SX * inv, -1.f, 1.f), cy = clampf(SY * inv, -1.f, 1.f);
        center[bi * 2] = cx; center[bi * 2 + 1] = cy;
        const float eps = 1e-6f;
        float ax = -log2f(clampf((cx + 1.f) * 0.5f, eps, 1.f - eps));
        float ay = -log2f(clampf((cy + 1.f) * 0.5f, eps, 1.f - eps));
        a_log[bi * 2] = ax;  a_log[bi * 2 + 1] = ay;
        inva[bi * 2] = 1.f / ax; inva[bi * 2 + 1] = 1.f / ay;
        float ncx = (cx + 1.f) * 0.5f, ncy = (cy + 1.f) * 0.5f;
        out_corr[bi * 2]     = (1.f + 2.f * (float)P_ / (float)WG_) * ncx - (float)P_ / (float)WG_;
        out_corr[bi * 2 + 1] = (1.f + 2.f * (float)P_ / (float)HG_) * ncy - (float)P_ / (float)HG_;
    }
}

// --------------------------------------------- pad z into channel-last image
__global__ __launch_bounds__(256) void k_zpad(const float* z, float* z_img) {
    size_t i = (size_t)blockIdx.x * 256 + threadIdx.x;
    if (i >= (size_t)B_ * NG_ * C_) return;
    int c = i % C_;
    size_t p = i / C_;
    int b = (int)(p / NG_), g = (int)(p % NG_);
    int y = g / WG_, x = g % WG_;
    float v = 0.f;
    if (y >= P_ && y < P_ + H_ && x >= P_ && x < P_ + W_)
        v = z[((size_t)b * N_ + (y - P_) * W_ + (x - P_)) * C_ + c];
    z_img[i] = v;
}

// --------------------------------------------------------- dense-grid PE
__global__ __launch_bounds__(256) void k_zpe(const float* G, float* z_pe_d) {
    int i = blockIdx.x * 256 + threadIdx.x;
    if (i >= NG_ * (C_ / 2)) return;
    int f = i % (C_ / 2), g = i / (C_ / 2);
    int y = g / WG_, x = g % WG_;
    float cx = 2.f * (((float)x + 0.5f) / (float)WG_) - 1.f;
    float cy = 2.f * (((float)y + 0.5f) / (float)HG_) - 1.f;
    float pr = (cx * G[f] + cy * G[(C_ / 2) + f]) * TWO_PI;
    z_pe_d[(size_t)g * C_ + f]            = sinf(pr);
    z_pe_d[(size_t)g * C_ + (C_ / 2) + f] = cosf(pr);
}

// --------------------------------------------- warp / reco grids per (b,i)
__global__ __launch_bounds__(256) void k_warp(const float* a_log, const float* inva,
                                              float* warp, float* reco) {
    int i = blockIdx.x * 256 + threadIdx.x;
    if (i >= BN_ * NG_) return;
    int bi = i / NG_, g = i % NG_;
    int y = g / WG_, x = g % WG_;
    const float eps = 1e-6f;
    float c01x = clampf((float)x / (float)(WG_ - 1), eps, 1.f);
    float c01y = clampf((float)y / (float)(HG_ - 1), eps, 1.f);
    float ax = a_log[bi * 2], ay = a_log[bi * 2 + 1];
    float ix = inva[bi * 2],  iy = inva[bi * 2 + 1];
    warp[(size_t)i * 2]     = clampf((powf(c01x, ax) - 0.5f) * 2.f, -1.f, 1.f);
    warp[(size_t)i * 2 + 1] = clampf((powf(c01y, ay) - 0.5f) * 2.f, -1.f, 1.f);
    reco[(size_t)i * 2]     = clampf((powf(c01x, ix) - 0.5f) * 2.f, -1.f, 1.f);
    reco[(size_t)i * 2 + 1] = clampf((powf(c01y, iy) - 0.5f) * 2.f, -1.f, 1.f);
}

// ---------------------------------------------------- wc_pe (B,n,n,C)
__global__ __launch_bounds__(128) void k_wcpe(const float* center, const float* inva,
                                              const float* G, float* wc_pe) {
    int bij = blockIdx.x;           // (b*8+i)*8 + j
    int bi = bij / NQ_, j = bij % NQ_, b = bi / NQ_;
    int f = threadIdx.x;
    const float eps = 1e-6f;
    float ncx = clampf((center[(b * NQ_ + j) * 2]     + 1.f) * 0.5f, eps, 1.f);
    float ncy = clampf((center[(b * NQ_ + j) * 2 + 1] + 1.f) * 0.5f, eps, 1.f);
    float wcx = clampf((powf(ncx, inva[bi * 2])     - 0.5f) * 2.f, -1.f, 1.f);
    float wcy = clampf((powf(ncy, inva[bi * 2 + 1]) - 0.5f) * 2.f, -1.f, 1.f);
    float pr = (wcx * G[f] + wcy * G[(C_ / 2) + f]) * TWO_PI;
    wc_pe[(size_t)bij * C_ + f]            = sinf(pr);
    wc_pe[(size_t)bij * C_ + (C_ / 2) + f] = cosf(pr);
}

// ------------------------------- bilinear weights helper (masked, clamped)
__device__ __forceinline__ void bilin(float wx, float wy, int Wd, int Hd,
                                      int idx[4], float w[4]) {
    float xf = (wx + 1.f) * 0.5f * (float)Wd - 0.5f;
    float yf = (wy + 1.f) * 0.5f * (float)Hd - 0.5f;
    float x0f = floorf(xf), y0f = floorf(yf);
    int x0 = (int)x0f, y0 = (int)y0f;
    float fx = xf - x0f, fy = yf - y0f;
    float ww[4] = { (1.f - fx) * (1.f - fy), fx * (1.f - fy), (1.f - fx) * fy, fx * fy };
    int xs[4] = { x0, x0 + 1, x0, x0 + 1 };
    int ys[4] = { y0, y0, y0 + 1, y0 + 1 };
    #pragma unroll
    for (int k = 0; k < 4; ++k) {
        bool ok = (xs[k] >= 0 && xs[k] < Wd && ys[k] >= 0 && ys[k] < Hd);
        int cx = min(max(xs[k], 0), Wd - 1), cy = min(max(ys[k], 0), Hd - 1);
        idx[k] = cy * Wd + cx;
        w[k] = ok ? ww[k] : 0.f;
    }
}

// --------------------------------- zw grid-sample -> zwf (rows x C) fp32
__global__ __launch_bounds__(256) void k_zw(const float* z_img, const float* warp,
                                            float* zwf) {
    int p = blockIdx.x;                 // 0..ROWS_-1
    int bi = p / NG_, b = bi / NQ_;
    int idx[4]; float w[4];
    bilin(warp[(size_t)p * 2], warp[(size_t)p * 2 + 1], WG_, HG_, idx, w);
    const float* base = z_img + (size_t)b * NG_ * C_;
    int c = threadIdx.x;
    float v = w[0] * base[(size_t)idx[0] * C_ + c] + w[1] * base[(size_t)idx[1] * C_ + c]
            + w[2] * base[(size_t)idx[2] * C_ + c] + w[3] * base[(size_t)idx[3] * C_ + c];
    zwf[(size_t)p * C_ + c] = v;
}

// --------------------------------- fw grid-sample: fw[b,i,j,g]
__global__ __launch_bounds__(256) void k_fw(const float* fixm, const float* warp,
                                            float* fw) {
    int p = blockIdx.x * 256 + threadIdx.x;
    if (p >= BN_ * NG_) return;
    int bi = p / NG_, b = bi / NQ_, g = p % NG_;
    int idx[4]; float w[4];
    bilin(warp[(size_t)p * 2], warp[(size_t)p * 2 + 1], WG_, HG_, idx, w);
    #pragma unroll
    for (int j = 0; j < NQ_; ++j) {
        const float* fm = fixm + (size_t)(b * NQ_ + j) * NG_;
        fw[((size_t)bi * NQ_ + j) * NG_ + g] =
            w[0] * fm[idx[0]] + w[1] * fm[idx[1]] + w[2] * fm[idx[2]] + w[3] * fm[idx[3]];
    }
}

// --------------------------------- q_w[b,i,j,c] = sum_g zwf * fw
__global__ __launch_bounds__(256) void k_qw(const float* zwf, const float* fw, float* q_w) {
    int bij = blockIdx.x;
    int bi = bij / NQ_, j = bij % NQ_;
    int c = threadIdx.x;
    const float* fwr = fw + ((size_t)bi * NQ_ + j) * NG_;
    const float* zr  = zwf + (size_t)bi * NG_ * C_;
    __shared__ float sf[256];
    float acc = 0.f;
    for (int g0 = 0; g0 < NG_; g0 += 256) {
        int gg = g0 + threadIdx.x;
        sf[threadIdx.x] = (gg < NG_) ? fwr[gg] : 0.f;
        __syncthreads();
        int lim = min(256, NG_ - g0);
        for (int k = 0; k < lim; ++k)
            acc += sf[k] * zr[(size_t)(g0 + k) * C_ + c];
        __syncthreads();
    }
    q_w[(size_t)bij * C_ + c] = acc;
}

// =================================================== WMMA: K/V projection
// outbf[M x 256] = bf16( (A [+ pe[row%Ng]]) @ W + bias )
template <bool HAS_PE>
__global__ __launch_bounds__(256) void k_proj_wmma(const float* __restrict__ Arows,
                                                   const float* __restrict__ pe,
                                                   const float* __restrict__ Wt,
                                                   const float* __restrict__ bias,
                                                   __bf16* __restrict__ outbf) {
    int mt = blockIdx.x;
    int wave = threadIdx.x >> 5, lane = threadIdx.x & 31, l15 = lane & 15;
    int row = mt * 16 + l15;
    const float* ar = Arows + (size_t)row * C_;
    const float* pr = HAS_PE ? (pe + (size_t)(row % NG_) * C_) : nullptr;
    int col0 = (wave * 2 + 0) * 16, col1 = (wave * 2 + 1) * 16;
    v8f acc0 = {0.f,0.f,0.f,0.f,0.f,0.f,0.f,0.f};
    v8f acc1 = {0.f,0.f,0.f,0.f,0.f,0.f,0.f,0.f};
    // prefetch next row-tile's A stream (memory-bound side of roofline)
    __builtin_prefetch(Arows + (size_t)(row + 16) * C_, 0, 1);
    for (int k0 = 0; k0 < C_; k0 += 32) {
        int kb = k0 + ((lane < 16) ? 0 : 8);
        const float* ap = ar + kb;
        float4 x0 = *(const float4*)(ap + 0);
        float4 x1 = *(const float4*)(ap + 4);
        float4 x2 = *(const float4*)(ap + 16);
        float4 x3 = *(const float4*)(ap + 20);
        if (HAS_PE) {
            const float* pp = pr + kb;
            float4 p0 = *(const float4*)(pp + 0);
            float4 p1 = *(const float4*)(pp + 4);
            float4 p2 = *(const float4*)(pp + 16);
            float4 p3 = *(const float4*)(pp + 20);
            x0.x += p0.x; x0.y += p0.y; x0.z += p0.z; x0.w += p0.w;
            x1.x += p1.x; x1.y += p1.y; x1.z += p1.z; x1.w += p1.w;
            x2.x += p2.x; x2.y += p2.y; x2.z += p2.z; x2.w += p2.w;
            x3.x += p3.x; x3.y += p3.y; x3.z += p3.z; x3.w += p3.w;
        }
        v16bf af;
        af[0]  = f2bf(x0.x); af[1]  = f2bf(x0.y); af[2]  = f2bf(x0.z); af[3]  = f2bf(x0.w);
        af[4]  = f2bf(x1.x); af[5]  = f2bf(x1.y); af[6]  = f2bf(x1.z); af[7]  = f2bf(x1.w);
        af[8]  = f2bf(x2.x); af[9]  = f2bf(x2.y); af[10] = f2bf(x2.z); af[11] = f2bf(x2.w);
        af[12] = f2bf(x3.x); af[13] = f2bf(x3.y); af[14] = f2bf(x3.z); af[15] = f2bf(x3.w);
        int kbB = k0 + ((lane < 16) ? 0 : 8);
        v16bf bf0, bf1;
        #pragma unroll
        for (int e = 0; e < 16; ++e) {
            const float* wr = Wt + (size_t)(kbB + ((e < 8) ? e : (e + 8))) * C_;
            bf0[e] = f2bf(wr[col0 + l15]);
            bf1[e] = f2bf(wr[col1 + l15]);
        }
        acc0 = __builtin_amdgcn_wmma_f32_16x16x32_bf16(false, af, false, bf0, (short)0, acc0, false, false);
        acc1 = __builtin_amdgcn_wmma_f32_16x16x32_bf16(false, af, false, bf1, (short)0, acc1, false, false);
    }
    int mbase = mt * 16 + ((lane < 16) ? 0 : 8);
    #pragma unroll
    for (int rr = 0; rr < 8; ++rr) {
        int m = mbase + rr;
        outbf[(size_t)m * C_ + col0 + l15] = f2bf(acc0[rr] + bias[col0 + l15]);
        outbf[(size_t)m * C_ + col1 + l15] = f2bf(acc1[rr] + bias[col1 + l15]);
    }
}

// =================================================== WMMA: scores = QK^T
__global__ __launch_bounds__(256) void k_scores_wmma(const float* __restrict__ qh,
                                                     const __bf16* __restrict__ khbf,
                                                     float* __restrict__ scores) {
    int bh = blockIdx.x;                  // (b*8+i)*8 + h
    int bi = bh / HEADS_, h = bh % HEADS_;
    int wave = threadIdx.x >> 5, lane = threadIdx.x & 31, l15 = lane & 15;
    int j = l15;
    v16bf af;
    int kb = (lane < 16) ? 0 : 16;
    #pragma unroll
    for (int e = 0; e < 16; ++e) {
        int d = kb + e;
        float v = (j < NQ_) ? qh[((size_t)bi * NQ_ + j) * C_ + h * HD_ + d] : 0.f;
        af[e] = f2bf(v);
    }
    const __bf16* kbase = khbf + (size_t)bi * NG_ * C_;
    const float scale = 0.17677669529663689f;   // 1/sqrt(32)
    int kbB = (lane < 16) ? 0 : 16;
    for (int gt = wave; gt < GT_; gt += 8) {
        int g0 = gt * 16;
        int gc = g0 + l15; if (gc >= NG_) gc = NG_ - 1;
        // 16 contiguous bf16 (32B, 32B-aligned): single vector load
        v16bf bfv = *(const v16bf*)(kbase + (size_t)gc * C_ + h * HD_ + kbB);
        v8f acc = {0.f,0.f,0.f,0.f,0.f,0.f,0.f,0.f};
        acc = __builtin_amdgcn_wmma_f32_16x16x32_bf16(false, af, false, bfv, (short)0, acc, false, false);
        int g = g0 + l15;
        if (g < NG_) {
            int jb = (lane < 16) ? 0 : 8;
            #pragma unroll
            for (int rr = 0; rr < 8; ++rr) {
                int jj = jb + rr;
                if (jj < NQ_)
                    scores[((size_t)bh * NQ_ + jj) * NG_ + g] = acc[rr] * scale;
            }
        }
    }
}

// ------------------------------------------------- row softmax over Ng
__global__ __launch_bounds__(256) void k_softmax(float* scores) {
    int r = blockIdx.x, tid = threadIdx.x;
    float* row = scores + (size_t)r * NG_;
    __shared__ float sred[256];
    float m = -1e30f;
    for (int i = tid; i < NG_; i += 256) m = fmaxf(m, row[i]);
    sred[tid] = m; __syncthreads();
    for (int s = 128; s > 0; s >>= 1) { if (tid < s) sred[tid] = fmaxf(sred[tid], sred[tid + s]); __syncthreads(); }
    float mx = sred[0]; __syncthreads();
    float sum = 0.f;
    for (int i = tid; i < NG_; i += 256) { float e = expf(row[i] - mx); row[i] = e; sum += e; }
    sred[tid] = sum; __syncthreads();
    for (int s = 128; s > 0; s >>= 1) { if (tid < s) sred[tid] += sred[tid + s]; __syncthreads(); }
    float inv = 1.f / sred[0];
    for (int i = tid; i < NG_; i += 256) row[i] *= inv;
}

// =================================================== WMMA: attn @ V
__global__ __launch_bounds__(64) void k_attnout_wmma(const float* __restrict__ attn,
                                                     const __bf16* __restrict__ vhbf,
                                                     float* __restrict__ attn_o) {
    int bh = blockIdx.x;
    int bi = bh / HEADS_, h = bh % HEADS_;
    int nt = threadIdx.x >> 5;            // N-tile 0/1 (d 0..15 / 16..31)
    int lane = threadIdx.x & 31, l15 = lane & 15;
    int j = l15;
    const float* arow = attn + (size_t)bh * NQ_ * NG_;
    const __bf16* vbase = vhbf + (size_t)bi * NG_ * C_;
    int dcol = h * HD_ + nt * 16 + l15;
    v8f acc = {0.f,0.f,0.f,0.f,0.f,0.f,0.f,0.f};
    int kb = (lane < 16) ? 0 : 16;
    int kbB = (lane < 16) ? 0 : 16;
    for (int gt = 0; gt < GT_; ++gt) {
        int g0 = gt * 16;
        v16bf af;
        #pragma unroll
        for (int e = 0; e < 16; ++e) {
            int gg = g0 + kb + e;
            float v = (j < NQ_ && gg < NG_) ? arow[(size_t)j * NG_ + gg] : 0.f;
            af[e] = f2bf(v);
        }
        v16bf bfv;
        #pragma unroll
        for (int e = 0; e < 16; ++e) {
            int gg = g0 + kbB + e; if (gg >= NG_) gg = NG_ - 1;
            bfv[e] = vbase[(size_t)gg * C_ + dcol];
        }
        acc = __builtin_amdgcn_wmma_f32_16x16x32_bf16(false, af, false, bfv, (short)0, acc, false, false);
    }
    int jb = (lane < 16) ? 0 : 8;
    #pragma unroll
    for (int rr = 0; rr < 8; ++rr) {
        int jj = jb + rr;
        if (jj < NQ_)
            attn_o[((size_t)bi * NQ_ + jj) * C_ + dcol] = acc[rr];
    }
}

// ------------------------------- layernorm(q_w + proj_o) + q_warp output
__global__ __launch_bounds__(256) void k_ln(const float* q_w, const float* proj_o,
                                            const float* gamma, const float* beta,
                                            const float* wc_pe, float* qn, float* out_qwarp) {
    int bij = blockIdx.x, tid = threadIdx.x;
    __shared__ float sred[256];
    float x = q_w[(size_t)bij * C_ + tid] + proj_o[(size_t)bij * C_ + tid];
    sred[tid] = x; __syncthreads();
    for (int s = 128; s > 0; s >>= 1) { if (tid < s) sred[tid] += sred[tid + s]; __syncthreads(); }
    float mean = sred[0] / (float)C_; __syncthreads();
    float d = x - mean;
    sred[tid] = d * d; __syncthreads();
    for (int s = 128; s > 0; s >>= 1) { if (tid < s) sred[tid] += sred[tid + s]; __syncthreads(); }
    float var = sred[0] / (float)C_;
    float qv = d * rsqrtf(var + 1e-5f) * gamma[tid] + beta[tid];
    qn[(size_t)bij * C_ + tid] = qv;
    out_qwarp[(size_t)bij * C_ + tid] = qv + wc_pe[(size_t)bij * C_ + tid];
}

__global__ __launch_bounds__(256) void k_diag(const float* qn, float* qnd) {
    int i = blockIdx.x * 256 + threadIdx.x;
    if (i >= BN_ * C_) return;
    int bi = i / C_, c = i % C_;
    qnd[i] = qn[((size_t)bi * NQ_ + (bi % NQ_)) * C_ + c];
}

__global__ __launch_bounds__(256) void k_iou(const float* q2d, const float* iou_w,
                                             const float* iou_b, float* out_iou) {
    int bi = blockIdx.x, tid = threadIdx.x;
    __shared__ float sred[256];
    sred[tid] = q2d[(size_t)bi * C_ + tid] * iou_w[tid];
    __syncthreads();
    for (int s = 128; s > 0; s >>= 1) { if (tid < s) sred[tid] += sred[tid + s]; __syncthreads(); }
    if (tid == 0) out_iou[bi] = sred[0] + iou_b[0];
}

__global__ __launch_bounds__(256) void k_mdiag(const float* __restrict__ q2d,
                                               const float* __restrict__ zwf,
                                               float* __restrict__ m_diag) {
    int p = blockIdx.x * 256 + threadIdx.x;
    if (p >= ROWS_) return;
    int bi = p / NG_;
    const float4* qr = (const float4*)(q2d + (size_t)bi * C_);
    const float4* zr = (const float4*)(zwf + (size_t)p * C_);
    float acc = 0.f;
    #pragma unroll 4
    for (int c = 0; c < C_ / 4; ++c) {
        float4 a = qr[c], b = zr[c];
        acc += a.x * b.x + a.y * b.y + a.z * b.z + a.w * b.w;
    }
    m_diag[p] = acc;
}

__global__ __launch_bounds__(256) void k_pred(const float* m_diag, const float* reco,
                                              float* out_pred) {
    int i = blockIdx.x * 256 + threadIdx.x;
    if (i >= BN_ * N_) return;
    int bi = i / N_, rem = i % N_;
    int yy = rem / W_, xx = rem % W_;
    int g = (yy + P_) * WG_ + (xx + P_);
    float rx = reco[((size_t)bi * NG_ + g) * 2];
    float ry = reco[((size_t)bi * NG_ + g) * 2 + 1];
    int idx[4]; float w[4];
    bilin(rx, ry, WG_, HG_, idx, w);
    const float* md = m_diag + (size_t)bi * NG_;
    out_pred[i] = w[0] * md[idx[0]] + w[1] * md[idx[1]] + w[2] * md[idx[2]] + w[3] * md[idx[3]];
}

// ============================================================== launcher
extern "C" void kernel_launch(void* const* d_in, const int* in_sizes, int n_in,
                              void* d_out, int out_size, void* d_ws, size_t ws_size,
                              hipStream_t stream) {
    (void)in_sizes; (void)n_in; (void)out_size; (void)ws_size;
    const float* q      = (const float*)d_in[0];
    const float* z      = (const float*)d_in[1];
    const float* q_pe   = (const float*)d_in[2];
    const float* z_pe   = (const float*)d_in[3];
    const float* fix_w1 = (const float*)d_in[4];
    const float* fix_b1 = (const float*)d_in[5];
    const float* fix_w2 = (const float*)d_in[6];
    const float* fix_b2 = (const float*)d_in[7];
    const float* mlp_w1 = (const float*)d_in[8];
    const float* mlp_b1 = (const float*)d_in[9];
    const float* mlp_w2 = (const float*)d_in[10];
    const float* mlp_b2 = (const float*)d_in[11];
    const float* norm_g = (const float*)d_in[12];
    const float* norm_b = (const float*)d_in[13];
    const float* iou_w  = (const float*)d_in[14];
    const float* iou_b  = (const float*)d_in[15];
    const float* wq     = (const float*)d_in[16];
    const float* bq     = (const float*)d_in[17];
    const float* wk     = (const float*)d_in[18];
    const float* bk     = (const float*)d_in[19];
    const float* wv     = (const float*)d_in[20];
    const float* bv     = (const float*)d_in[21];
    const float* wo     = (const float*)d_in[22];
    const float* bo     = (const float*)d_in[23];
    const float* pe_g   = (const float*)d_in[24];

    float* out = (float*)d_out;
    float* out_qwarp = out;                       // (B,n,n,C)   32768
    float* out_pred  = out + 32768;               // (B,n,80,80) 102400
    float* out_iou   = out + 32768 + 102400;      // (B,n,1)     16
    float* out_fixm  = out + 32768 + 102400 + 16; // (B,n,80,80) 102400
    float* out_corr  = out + 32768 + 102400 + 16 + 102400; // (B,n,2) 32

    float* ws = (float*)d_ws;
    size_t o = 0;
    float* t_q    = ws + o; o += (size_t)BN_ * C_;
    float* fixh   = ws + o; o += (size_t)BN_ * HID_;
    float* fixv   = ws + o; o += (size_t)BN_ * C_;
    float* fixm   = ws + o; o += (size_t)BN_ * NG_;
    float* center = ws + o; o += 32;
    float* a_log  = ws + o; o += 32;
    float* inva   = ws + o; o += 32;
    float* z_img  = ws + o; o += (size_t)B_ * NG_ * C_;
    float* z_pe_d = ws + o; o += (size_t)NG_ * C_;
    float* warp   = ws + o; o += (size_t)BN_ * NG_ * 2;
    float* reco   = ws + o; o += (size_t)BN_ * NG_ * 2;
    float* wc_pe  = ws + o; o += (size_t)BN_ * NQ_ * C_;
    float* zwf    = ws + o; o += (size_t)ROWS_ * C_;
    float* fw     = ws + o; o += (size_t)BN_ * NQ_ * NG_;
    float* q_w    = ws + o; o += (size_t)BN_ * NQ_ * C_;
    float* qh     = ws + o; o += (size_t)BN_ * NQ_ * C_;
    float* scores = ws + o; o += (size_t)BN_ * HEADS_ * NQ_ * NG_;
    float* attn_o = ws + o; o += (size_t)BN_ * NQ_ * C_;
    float* proj_o = ws + o; o += (size_t)BN_ * NQ_ * C_;
    float* qn     = ws + o; o += (size_t)BN_ * NQ_ * C_;
    float* qnd    = ws + o; o += (size_t)BN_ * C_;
    float* h2     = ws + o; o += (size_t)BN_ * HID_;
    float* q2d    = ws + o; o += (size_t)BN_ * C_;
    float* m_diag = ws + o; o += (size_t)BN_ * NG_;
    __bf16* khbf = (__bf16*)(ws + o); o += (size_t)ROWS_ * C_ / 2;
    __bf16* vhbf = (__bf16*)(ws + o);

    // 1) fix MLP
    hipLaunchKernelGGL(k_add, dim3((BN_ * C_ + 255) / 256), dim3(256), 0, stream,
                       q, q_pe, t_q, BN_ * C_);
    hipLaunchKernelGGL(k_sgemm, dim3(BN_), dim3(256), C_ * 4, stream,
                       t_q, (const float*)nullptr, fix_w1, fix_b1, fixh, C_, HID_, 1);
    hipLaunchKernelGGL(k_sgemm, dim3(BN_), dim3(256), HID_ * 4, stream,
                       fixh, (const float*)nullptr, fix_w2, fix_b2, fixv, HID_, C_, 0);

    // 2) logits/softmax -> fixm (padded), center/a/corr
    hipLaunchKernelGGL(k_fill0, dim3((BN_ * NG_ + 255) / 256), dim3(256), 0, stream,
                       fixm, BN_ * NG_);
    hipLaunchKernelGGL(k_fixm, dim3(BN_), dim3(256), 0, stream,
                       fixv, z, z_pe, fixm, out_fixm, center, a_log, inva, out_corr);

    // 3) padded image / dense PE / warp grids / wc_pe
    hipLaunchKernelGGL(k_zpad, dim3((unsigned)(((size_t)B_ * NG_ * C_ + 255) / 256)), dim3(256), 0, stream,
                       z, z_img);
    hipLaunchKernelGGL(k_zpe, dim3((NG_ * (C_ / 2) + 255) / 256), dim3(256), 0, stream,
                       pe_g, z_pe_d);
    hipLaunchKernelGGL(k_warp, dim3((BN_ * NG_ + 255) / 256), dim3(256), 0, stream,
                       a_log, inva, warp, reco);
    hipLaunchKernelGGL(k_wcpe, dim3(BN_ * NQ_), dim3(128), 0, stream,
                       center, inva, pe_g, wc_pe);

    // 4) grid-samples and q_w
    hipLaunchKernelGGL(k_zw, dim3(ROWS_), dim3(256), 0, stream, z_img, warp, zwf);
    hipLaunchKernelGGL(k_fw, dim3((BN_ * NG_ + 255) / 256), dim3(256), 0, stream,
                       fixm, warp, fw);
    hipLaunchKernelGGL(k_qw, dim3(BN_ * NQ_), dim3(256), 0, stream, zwf, fw, q_w);

    // 5) WMMA K/V projections (M=141376, N=K=256)
    hipLaunchKernelGGL((k_proj_wmma<true>), dim3(ROWS_ / 16), dim3(256), 0, stream,
                       zwf, z_pe_d, wk, bk, khbf);
    hipLaunchKernelGGL((k_proj_wmma<false>), dim3(ROWS_ / 16), dim3(256), 0, stream,
                       zwf, (const float*)nullptr, wv, bv, vhbf);

    // 6) Q projection (small) then WMMA attention
    hipLaunchKernelGGL(k_sgemm, dim3(BN_ * NQ_), dim3(256), C_ * 4, stream,
                       q_w, wc_pe, wq, bq, qh, C_, C_, 0);
    hipLaunchKernelGGL(k_scores_wmma, dim3(BN_ * HEADS_), dim3(256), 0, stream,
                       qh, khbf, scores);
    hipLaunchKernelGGL(k_softmax, dim3(BN_ * HEADS_ * NQ_), dim3(256), 0, stream, scores);
    hipLaunchKernelGGL(k_attnout_wmma, dim3(BN_ * HEADS_), dim3(64), 0, stream,
                       scores, vhbf, attn_o);
    hipLaunchKernelGGL(k_sgemm, dim3(BN_ * NQ_), dim3(256), C_ * 4, stream,
                       attn_o, (const float*)nullptr, wo, bo, proj_o, C_, C_, 0);

    // 7) layernorm + q_warp output, diag MLP, iou
    hipLaunchKernelGGL(k_ln, dim3(BN_ * NQ_), dim3(256), 0, stream,
                       q_w, proj_o, norm_g, norm_b, wc_pe, qn, out_qwarp);
    hipLaunchKernelGGL(k_diag, dim3((BN_ * C_ + 255) / 256), dim3(256), 0, stream, qn, qnd);
    hipLaunchKernelGGL(k_sgemm, dim3(BN_), dim3(256), C_ * 4, stream,
                       qnd, (const float*)nullptr, mlp_w1, mlp_b1, h2, C_, HID_, 1);
    hipLaunchKernelGGL(k_sgemm, dim3(BN_), dim3(256), HID_ * 4, stream,
                       h2, (const float*)nullptr, mlp_w2, mlp_b2, q2d, HID_, C_, 0);
    hipLaunchKernelGGL(k_iou, dim3(BN_), dim3(256), 0, stream, q2d, iou_w, iou_b, out_iou);

    // 8) mask prediction
    hipLaunchKernelGGL(k_mdiag, dim3((ROWS_ + 255) / 256), dim3(256), 0, stream,
                       q2d, zwf, m_diag);
    hipLaunchKernelGGL(k_pred, dim3((BN_ * N_ + 255) / 256), dim3(256), 0, stream,
                       m_diag, reco, out_pred);
}
